// Attention_70523363000802
// MI455X (gfx1250) — matrix operand order repaired and checked by reading, not compiled, not run on hardware
//
#include <hip/hip_runtime.h>
#include <hip/hip_bf16.h>
#include <hip/hip_fp16.h>

typedef __attribute__((ext_vector_type(16))) _Float16 v16h;
typedef __attribute__((ext_vector_type(8)))  float    v8f;

#define BATCH 4096
#define CC    32      // channels (= WMMA K for the main GEMM)
#define P     196     // spatial positions
#define PPAD  208     // 13 * 16
#define ED    512     // encoder dim
#define AD    128     // attention dim

// tanh(x) = 1 - 2/(exp(2x)+1): exact identity, saturates to +/-1 through
// __expf overflow/underflow; 1 TRANS (v_exp_f32) + ~3 VALU ops instead of
// the long branchy libm tanhf expansion.
__device__ __forceinline__ float fast_tanh(float x) {
    const float e = __expf(2.f * x);
    return 1.f - 2.f / (e + 1.f);
}

// ---------------------------------------------------------------------------
// Kernel 1: U_h = prev @ U_w^T + U_b  (WMMA, K=512 in 16 steps)
//           beta = sigmoid(prev @ fb_w^T + fb_b)
// One block handles 16 batch rows (one 16-row M tile). 4 waves:
//   wave w -> U_h N-tiles {w, w+4};  waves 0,1 -> fb N-tiles {0,1}.
// ---------------------------------------------------------------------------
__launch_bounds__(128)
__global__ void uh_beta_kernel(const float* __restrict__ prev,  // [B,512]
                               const float* __restrict__ Uw,    // [128,512]
                               const float* __restrict__ Ub,    // [128]
                               const float* __restrict__ fbw,   // [32,512]
                               const float* __restrict__ fbb,   // [32]
                               float* __restrict__ uh,          // [B,128]
                               float* __restrict__ beta)        // [B,32]
{
    const int tid  = threadIdx.x;
    const int lane = tid & 31;
    const int wave = tid >> 5;
    const int half = lane >> 4;
    const int ln   = lane & 15;
    const int bbase = blockIdx.x * 16;

    __shared__ _Float16 sPrev[16 * 522];   // [m][k], pitch 522 (conflict-free A reads)
    __shared__ _Float16 sUw[128 * 34];     // [a][kk], pitch 34 (17 dwords, coprime w/ 64)
    __shared__ _Float16 sFb[32 * 34];      // [c][kk]

    // stage prev_hidden 16x512 (f32 -> f16)
    for (int i = tid; i < 16 * ED; i += 128) {
        int r = i >> 9, k = i & 511;
        sPrev[r * 522 + k] = (_Float16)prev[(size_t)(bbase + r) * ED + k];
    }

    v8f accU0 = {}, accU1 = {}, accF = {};
    const int n0 = wave * 16;
    const int n1 = (wave + 4) * 16;

    for (int ks = 0; ks < 16; ++ks) {
        __syncthreads();
        // stage U_w K-chunk [128][32]
        for (int i = tid; i < 128 * 32; i += 128) {
            int a = i >> 5, kk = i & 31;
            sUw[a * 34 + kk] = (_Float16)Uw[(size_t)a * ED + ks * 32 + kk];
        }
        // stage fb_w K-chunk [32][32]
        for (int i = tid; i < 32 * 32; i += 128) {
            int c = i >> 5, kk = i & 31;
            sFb[c * 34 + kk] = (_Float16)fbw[(size_t)c * ED + ks * 32 + kk];
        }
        __syncthreads();

        // A fragment: row m = lane&15; k = i + half*16
        v16h a;
        {
            const _Float16* sp = &sPrev[ln * 522 + ks * 32 + half * 16];
            #pragma unroll
            for (int i = 0; i < 8; ++i)  a[i] = sp[i];
            #pragma unroll
            for (int i = 8; i < 16; ++i) a[i] = sp[i];
        }
        // B fragments: col n = lane&15; k = i + half*16 (consecutive halfs)
        v16h bu0, bu1;
        {
            const _Float16* sb0 = &sUw[(n0 + ln) * 34 + half * 16];
            const _Float16* sb1 = &sUw[(n1 + ln) * 34 + half * 16];
            #pragma unroll
            for (int i = 0; i < 16; ++i) { bu0[i] = sb0[i]; bu1[i] = sb1[i]; }
        }
        accU0 = __builtin_amdgcn_wmma_f32_16x16x32_f16(false, a, false, bu0,
                                                       (short)0, accU0, false, false);
        accU1 = __builtin_amdgcn_wmma_f32_16x16x32_f16(false, a, false, bu1,
                                                       (short)0, accU1, false, false);
        if (wave < 2) {   // wave-uniform branch: EXEC stays all-1s for WMMA
            v16h bfr;
            const _Float16* sb = &sFb[(wave * 16 + ln) * 34 + half * 16];
            #pragma unroll
            for (int i = 0; i < 16; ++i) bfr[i] = sb[i];
            accF = __builtin_amdgcn_wmma_f32_16x16x32_f16(false, a, false, bfr,
                                                          (short)0, accF, false, false);
        }
    }

    // C/D layout: element (m = j + 8*half, n = lane&15) in accX[j]
    #pragma unroll
    for (int j = 0; j < 8; ++j) {
        const int m = j + half * 8;
        uh[(size_t)(bbase + m) * AD + n0 + ln] = accU0[j] + Ub[n0 + ln];
        uh[(size_t)(bbase + m) * AD + n1 + ln] = accU1[j] + Ub[n1 + ln];
    }
    if (wave < 2) {
        const int cb = wave * 16 + ln;
        #pragma unroll
        for (int j = 0; j < 8; ++j) {
            const int m = j + half * 8;
            const float x = accF[j] + fbb[cb];
            beta[(size_t)(bbase + m) * CC + cb] = 1.f / (1.f + __expf(-x));
        }
    }
}

// ---------------------------------------------------------------------------
// Kernel 2: per batch element — fused attention:
//   W_s = img_parts @ W_w^T (WMMA, single K=32 step per tile)
//   score = tanh(W_s + U_h + W_b) . v ; softmax ; context = att^T @ img * beta
// Block = 128 threads (4 waves); one block per batch element.
// ---------------------------------------------------------------------------
__launch_bounds__(128)
__global__ void att_kernel(const float* __restrict__ img,   // [B,32,196]
                           const float* __restrict__ Ww,    // [128,32]
                           const float* __restrict__ Wb,    // [128]
                           const float* __restrict__ vw,    // [128]
                           const float* __restrict__ vb,    // [1]
                           const float* __restrict__ uh,    // [B,128]
                           const float* __restrict__ beta,  // [B,32]
                           float* __restrict__ out_ctx,     // [B,32]
                           float* __restrict__ out_att)     // [B,196]
{
    const int tid  = threadIdx.x;
    const int lane = tid & 31;
    const int wave = tid >> 5;
    const int half = lane >> 4;
    const int ln   = lane & 15;
    const int b    = blockIdx.x;

    __shared__ _Float16 sImg[CC * PPAD];  // [c][p]  (A reads: lanes hit consecutive halfs)
    __shared__ _Float16 sW[AD * 34];      // [a][c]  pitch 34
    __shared__ float sUhW[AD];            // U_h(b) + W_b
    __shared__ float sV[AD];
    __shared__ float sScore[PPAD];
    __shared__ float sRed[128];
    __shared__ float sCtx[128];

    // zero the spatial padding (p in [196,208))
    for (int i = tid; i < CC * (PPAD - P); i += 128) {
        int c = i / (PPAD - P), p = P + i % (PPAD - P);
        sImg[c * PPAD + p] = (_Float16)0.f;
    }
    // stage img tile [32][196], f32 -> f16 (read once from HBM, reused twice)
    const float* imgb = img + (size_t)b * CC * P;
    for (int i = tid; i < CC * P; i += 128) {
        int c = i / P, p = i % P;
        sImg[c * PPAD + p] = (_Float16)imgb[i];
    }
    // stage W_w [128][32]
    for (int i = tid; i < AD * CC; i += 128) {
        int a2 = i >> 5, c = i & 31;
        sW[a2 * 34 + c] = (_Float16)Ww[i];
    }
    for (int i = tid; i < AD; i += 128) {
        sUhW[i] = uh[(size_t)b * AD + i] + Wb[i];
        sV[i]   = vw[i];
    }
    __syncthreads();

    // hoist all 8 W_w B-fragments (K = full 32 channels)
    v16h bf[8];
    #pragma unroll
    for (int nt = 0; nt < 8; ++nt) {
        const _Float16* sb = &sW[(nt * 16 + ln) * 34 + half * 16];
        #pragma unroll
        for (int i = 0; i < 16; ++i) bf[nt][i] = sb[i];
    }

    const float vbias = vb[0];
    // 13 M-tiles of 16 positions, round-robin over 4 waves (wave-uniform loop)
    for (int mt = wave; mt < 13; mt += 4) {
        const int p0 = mt * 16;
        v16h a;
        {
            const int p = p0 + ln;
            const _Float16* sp = &sImg[(half * 16) * PPAD + p];
            #pragma unroll
            for (int i = 0; i < 8; ++i)  a[i] = sp[i * PPAD];
            #pragma unroll
            for (int i = 8; i < 16; ++i) a[i] = sp[i * PPAD];
        }
        float part[8] = {0.f, 0.f, 0.f, 0.f, 0.f, 0.f, 0.f, 0.f};
        #pragma unroll
        for (int nt = 0; nt < 8; ++nt) {
            v8f acc = {};
            acc = __builtin_amdgcn_wmma_f32_16x16x32_f16(false, a, false, bf[nt],
                                                         (short)0, acc, false, false);
            const int n  = nt * 16 + ln;
            const float u  = sUhW[n];
            const float vv = sV[n];
            #pragma unroll
            for (int j = 0; j < 8; ++j)
                part[j] += fast_tanh(acc[j] + u) * vv;   // fused tanh + v-dot epilogue
        }
        // reduce over n (lanes within each 16-lane half; masks < 16 stay in-half)
        #pragma unroll
        for (int j = 0; j < 8; ++j) {
            float s = part[j];
            s += __shfl_xor(s, 8, 32);
            s += __shfl_xor(s, 4, 32);
            s += __shfl_xor(s, 2, 32);
            s += __shfl_xor(s, 1, 32);
            if (ln == 0) {
                const int p = p0 + j + half * 8;
                if (p < P) sScore[p] = s + vbias;
            }
        }
    }
    __syncthreads();

    // ---- softmax over 196 positions ----
    float lmax = -3.4e38f;
    for (int p = tid; p < P; p += 128) lmax = fmaxf(lmax, sScore[p]);
    sRed[tid] = lmax; __syncthreads();
    for (int s = 64; s > 0; s >>= 1) {
        if (tid < s) sRed[tid] = fmaxf(sRed[tid], sRed[tid + s]);
        __syncthreads();
    }
    const float mx = sRed[0];
    __syncthreads();

    float lsum = 0.f;
    for (int p = tid; p < P; p += 128) {
        const float e = __expf(sScore[p] - mx);
        sScore[p] = e;
        lsum += e;
    }
    sRed[tid] = lsum; __syncthreads();
    for (int s = 64; s > 0; s >>= 1) {
        if (tid < s) sRed[tid] += sRed[tid + s];
        __syncthreads();
    }
    const float inv = 1.f / sRed[0];
    __syncthreads();

    for (int p = tid; p < P; p += 128) {
        const float w = sScore[p] * inv;
        sScore[p] = w;
        out_att[(size_t)b * P + p] = w;
    }
    __syncthreads();

    // ---- context = att^T @ img_parts, scaled by beta ----
    const int c = tid & 31;
    const int q = tid >> 5;                 // 196 = 4 * 49
    float acc = 0.f;
    for (int p = q * 49; p < q * 49 + 49; ++p)
        acc += (float)sImg[c * PPAD + p] * sScore[p];
    sCtx[tid] = acc;
    __syncthreads();
    if (tid < 32) {
        const float v = sCtx[tid] + sCtx[tid + 32] + sCtx[tid + 64] + sCtx[tid + 96];
        out_ctx[(size_t)b * CC + tid] = v * beta[(size_t)b * CC + tid];
    }
}

// ---------------------------------------------------------------------------
extern "C" void kernel_launch(void* const* d_in, const int* in_sizes, int n_in,
                              void* d_out, int out_size, void* d_ws, size_t ws_size,
                              hipStream_t stream) {
    const float* img  = (const float*)d_in[0];
    const float* prev = (const float*)d_in[1];
    const float* Ww   = (const float*)d_in[2];
    const float* Wb   = (const float*)d_in[3];
    const float* Uw   = (const float*)d_in[4];
    const float* Ub   = (const float*)d_in[5];
    const float* vw   = (const float*)d_in[6];
    const float* vb   = (const float*)d_in[7];
    const float* fbw  = (const float*)d_in[8];
    const float* fbb  = (const float*)d_in[9];

    // workspace: U_h [B,128] then beta [B,32]  (~2.5 MB)
    float* uh   = (float*)d_ws;
    float* betp = uh + (size_t)BATCH * AD;

    float* out_ctx = (float*)d_out;                       // [B,32]
    float* out_att = out_ctx + (size_t)BATCH * CC;        // [B,196]

    uh_beta_kernel<<<BATCH / 16, 128, 0, stream>>>(prev, Uw, Ub, fbw, fbb, uh, betp);
    att_kernel<<<BATCH, 128, 0, stream>>>(img, Ww, Wb, vw, vb, uh, betp,
                                          out_ctx, out_att);
}
